// FlashHSAInnerXHierarchicalSparseAttention_49237505081458
// MI455X (gfx1250) — matrix operand order repaired and checked by reading, not compiled, not run on hardware
//
#include <hip/hip_runtime.h>
#include <hip/hip_bf16.h>

// ---------------------------------------------------------------------------
// MI455X (gfx1250): compute-bound attention block (~65 GFLOP vs ~75 MB traffic)
// -> all MACs on v_wmma_f32_16x16x32_f16 (wave32), A/K/V tile staging as raw f16
// byte movement through GLOBAL_LOAD_ASYNC_TO_LDS_B128 (ASYNCcnt-tracked).
// ---------------------------------------------------------------------------

typedef _Float16 v16h __attribute__((ext_vector_type(16)));
typedef float    v8f  __attribute__((ext_vector_type(8)));

#define S_LEN 2048
#define HDIM  64

// ---- gfx1250 async global->LDS path (guarded; safe fallback) --------------
#ifdef __has_builtin
#  if __has_builtin(__builtin_amdgcn_global_load_async_to_lds_b128)
#    define HAVE_ASYNC_LDS 1
#  endif
#  if __has_builtin(__builtin_amdgcn_s_wait_asynccnt)
#    define HAVE_WAIT_ASYNC 1
#  endif
#endif
#ifndef HAVE_ASYNC_LDS
#  define HAVE_ASYNC_LDS 0
#endif

#if HAVE_ASYNC_LDS
// Builtin signature (from clang diagnostic): param0 = v4i addrspace(1)* (global),
// param1 = v4i addrspace(3)* (LDS), then Ii offset, Ii cpol.
typedef int v4i_vs __attribute__((__vector_size__(16)));
typedef __attribute__((address_space(1))) v4i_vs* gptr_b128;
typedef __attribute__((address_space(3))) v4i_vs* lptr_b128;
__device__ __forceinline__ void g2l_async_b128(const void* g, void* l) {
    __builtin_amdgcn_global_load_async_to_lds_b128(
        (gptr_b128)(unsigned long long)g,
        (lptr_b128)(unsigned int)(unsigned long long)l, 0, 0);
}
#  ifdef HAVE_WAIT_ASYNC
#    define WAIT_ASYNC0() __builtin_amdgcn_s_wait_asynccnt(0)
#  else
#    define WAIT_ASYNC0() asm volatile("s_wait_asynccnt 0x0" ::: "memory")
#  endif
#endif

// ---------------------------------------------------------------------------
// fp32 -> f16 bulk convert (one pass; makes every later stage pure f16 bytes)
// ---------------------------------------------------------------------------
__global__ __launch_bounds__(256) void cvt_f32_f16(
    const float* __restrict__ in, _Float16* __restrict__ out, int n4)
{
    const int i = blockIdx.x * 256 + threadIdx.x;
    if (i < n4) {
        const float4 f = ((const float4*)in)[i];
        _Float16 h4[4] = {(_Float16)f.x, (_Float16)f.y, (_Float16)f.z, (_Float16)f.w};
        ((uint2*)out)[i] = *(const uint2*)h4;
    }
}

// ---------------------------------------------------------------------------
// f16 GEMM, f32 WMMA accumulate. 256 threads = 8 waves. Block tile 128x128,
// K-step 32. Wave (wm,wn) computes 32x64 => 8 WMMAs/K-step, 4-way A reuse.
// LDS double-buffered; A tile staged with async global->LDS copies, B tile
// register-staged + transposed so K is contiguous for the fragment gather.
// C written as f16 (c_half=1, feeds next WMMA stage) or f32 (final output).
// ---------------------------------------------------------------------------
#define PA 40   // sA row pad (f16): 80B rows -> 16B-aligned b128 chunks, odd bank stride
#define PB 36   // sB row pad (f16)

__global__ __launch_bounds__(256) void gemm_h_wmma(
    const _Float16* __restrict__ A, int lda,
    const _Float16* __restrict__ B, int ldb,
    void* __restrict__ Cp, int ldc, int K, int c_half)
{
    __shared__ __align__(16) _Float16 sA[2][128][PA];   // [m][k]
    __shared__ __align__(16) _Float16 sB[2][128][PB];   // [n][k] (transposed)

    const int tid = threadIdx.x, lane = tid & 31, wave = tid >> 5;
    const int wm = wave & 3, wn = wave >> 2;
    const int lg = lane >> 4, l15 = lane & 15;
    const int bm = blockIdx.y * 128, bn = blockIdx.x * 128;

    v8f acc[2][4];
    #pragma unroll
    for (int a2 = 0; a2 < 2; ++a2)
        #pragma unroll
        for (int t = 0; t < 4; ++t)
            #pragma unroll
            for (int r = 0; r < 8; ++r) acc[a2][t][r] = 0.0f;

    uint4 rb[2];

    // A tile 128x32 f16 = 512 16B chunks; 2 per thread.
    auto stageA = [&](int k0, int buf) {
        #pragma unroll
        for (int i = 0; i < 2; ++i) {
            const int flat = i * 256 + tid;
            const int row = flat >> 2, kq = flat & 3;
            const _Float16* g = A + (size_t)(bm + row) * lda + k0 + kq * 8;
#if HAVE_ASYNC_LDS
            g2l_async_b128(g, &sA[buf][row][kq * 8]);
#else
            *(uint4*)&sA[buf][row][kq * 8] = *(const uint4*)g;
#endif
        }
    };
    // B tile 32x128 f16: 512 8-f16 chunks; 2 per thread (batched loads).
    auto loadB = [&](int k0) {
        #pragma unroll
        for (int i = 0; i < 2; ++i) {
            const int flat = i * 256 + tid;
            const int kk = flat >> 4, n0 = (flat & 15) * 8;
            rb[i] = *(const uint4*)(B + (size_t)(k0 + kk) * ldb + bn + n0);
        }
    };
    auto storeB = [&](int buf) {
        #pragma unroll
        for (int i = 0; i < 2; ++i) {
            const int flat = i * 256 + tid;
            const int kk = flat >> 4, n0 = (flat & 15) * 8;
            const _Float16* h = (const _Float16*)&rb[i];
            #pragma unroll
            for (int j = 0; j < 8; ++j) sB[buf][n0 + j][kk] = h[j];
        }
    };

    stageA(0, 0);
    loadB(0);
    storeB(0);
#if HAVE_ASYNC_LDS
    WAIT_ASYNC0();
#endif
    __syncthreads();

    int cur = 0;
    for (int k0 = 0; k0 < K; k0 += 32) {
        const int nxt = cur ^ 1;
        const bool more = (k0 + 32) < K;
        if (more) { stageA(k0 + 32, nxt); loadB(k0 + 32); }  // in flight over compute

        // A fragments (16x32, contiguous k = 16*lg + s), 2 per wave
        v16h afr[2];
        #pragma unroll
        for (int a2 = 0; a2 < 2; ++a2) {
            const int row = wm * 32 + a2 * 16 + l15;
            #pragma unroll
            for (int j = 0; j < 8; ++j) {
                const int kk = lg * 16 + 2 * j;
                afr[a2][2 * j]     = sA[cur][row][kk];
                afr[a2][2 * j + 1] = sA[cur][row][kk + 1];
            }
        }
        #pragma unroll
        for (int t = 0; t < 4; ++t) {
            v16h bfr;
            const int col = wn * 64 + t * 16 + l15;
            #pragma unroll
            for (int j = 0; j < 8; ++j) {
                const int kk = lg * 16 + 2 * j;
                bfr[2 * j]     = sB[cur][col][kk];
                bfr[2 * j + 1] = sB[cur][col][kk + 1];
            }
            acc[0][t] = __builtin_amdgcn_wmma_f32_16x16x32_f16(false, afr[0], false, bfr, (short)0, acc[0][t], false, false);
            acc[1][t] = __builtin_amdgcn_wmma_f32_16x16x32_f16(false, afr[1], false, bfr, (short)0, acc[1][t], false, false);
        }

        if (more) {
            storeB(nxt);
#if HAVE_ASYNC_LDS
            WAIT_ASYNC0();
#endif
        }
        __syncthreads();
        cur = nxt;
    }

    #pragma unroll
    for (int a2 = 0; a2 < 2; ++a2)
        #pragma unroll
        for (int t = 0; t < 4; ++t)
            #pragma unroll
            for (int r = 0; r < 8; ++r) {
                const int m = bm + wm * 32 + a2 * 16 + ((lane < 16) ? r : r + 8);
                const int n = bn + wn * 64 + t * 16 + l15;
                if (c_half) ((_Float16*)Cp)[(size_t)m * ldc + n] = (_Float16)acc[a2][t][r];
                else        ((float*)Cp)[(size_t)m * ldc + n]    = acc[a2][t][r];
            }
}

// ---------------------------------------------------------------------------
// RMSNorm (+ optional RoPE) in-place on f16 head vectors; stats in f32.
// One wave per (token, head); lane d owns RoPE pair (d, d+32).
// ---------------------------------------------------------------------------
__global__ __launch_bounds__(128) void norm_rope_h(
    _Float16* __restrict__ buf, const float* __restrict__ w,
    const int* __restrict__ positions, int nheads, int do_rope)
{
    const int gid  = blockIdx.x * 4 + (threadIdx.x >> 5);
    const int lane = threadIdx.x & 31;
    const int s = gid / nheads;
    const int h = gid - s * nheads;

    _Float16* p = buf + ((size_t)s * nheads + h) * HDIM;
    float x1 = (float)p[lane];
    float x2 = (float)p[lane + 32];
    float ss = x1 * x1 + x2 * x2;
    #pragma unroll
    for (int m = 16; m; m >>= 1) ss += __shfl_xor(ss, m, 32);
    const float r = rsqrtf(ss * (1.0f / 64.0f) + 1e-6f);
    x1 = x1 * r * w[lane];
    x2 = x2 * r * w[lane + 32];

    if (do_rope) {
        const float pos = (float)positions[s];
        const float inv = __expf(-(float)lane * (0.03125f * __logf(1.0e6f)));  // theta^(-d/32)
        const float ang = pos * inv;
        const float c = __cosf(ang), sn = __sinf(ang);
        const float o1 = x1 * c - x2 * sn;
        const float o2 = x2 * c + x1 * sn;
        x1 = o1; x2 = o2;
    }
    p[lane] = (_Float16)x1;
    p[lane + 32] = (_Float16)x2;
}

// ---------------------------------------------------------------------------
// Flash attention, f16 WMMA, f16 in / f16 out. One wave per (head, 16-row
// q-tile); 32-key tiles. K/V rows (128B) async-copied straight into LDS
// (wave-private region -> s_wait_asynccnt only, no barrier). Per tile:
// 4 WMMAs scores + masked online softmax + P bounce (f16) + 4 WMMAs PV.
// ---------------------------------------------------------------------------
#define PK 72   // sK/sV row pad (f16): 144B rows, 16B-aligned chunks
#define PP 40

__global__ __launch_bounds__(128) void attn_h_wmma(
    const _Float16* __restrict__ qp, const _Float16* __restrict__ kp,
    const _Float16* __restrict__ vp, _Float16* __restrict__ out,
    int hq, int hk, int window, int col_off, int out_ld)
{
    __shared__ __align__(16) _Float16 sK[4][32][PK];   // [wave][key][d]
    __shared__ __align__(16) _Float16 sV[4][32][PK];   // [wave][key][d]
    __shared__ __align__(16) _Float16 sP[4][16][PP];   // [wave][qrow][key]

    const int wave = threadIdx.x >> 5, lane = threadIdx.x & 31;
    const int lg = lane >> 4, l15 = lane & 15;
    const int gid = blockIdx.x * 4 + wave;
    const int h   = gid >> 7;          // S/16 = 128 q tiles
    const int qb  = (gid & 127) * 16;
    const int hkv = h / (hq / hk);

    // Q fragments (d-chunks [0,32) and [32,64)), pre-scaled by D^-0.5 = 0.125
    v16h qa0, qa1;
    {
        const _Float16* qrow = qp + ((size_t)(qb + l15) * hq + h) * HDIM;
        const _Float16 sc = (_Float16)0.125f;
        #pragma unroll
        for (int j = 0; j < 8; ++j) {
            const int kk = lg * 16 + 2 * j;
            qa0[2 * j]     = qrow[kk] * sc;
            qa0[2 * j + 1] = qrow[kk + 1] * sc;
            qa1[2 * j]     = qrow[32 + kk] * sc;
            qa1[2 * j + 1] = qrow[32 + kk + 1] * sc;
        }
    }

    v8f o[4];
    #pragma unroll
    for (int t = 0; t < 4; ++t)
        #pragma unroll
        for (int r = 0; r < 8; ++r) o[t][r] = 0.0f;
    float mrow[8], lrow[8];
    #pragma unroll
    for (int r = 0; r < 8; ++r) { mrow[r] = -1e30f; lrow[r] = 0.0f; }

    int kb_lo = 0;
    if (window > 0) { int lo = qb - (window - 1); if (lo > 0) kb_lo = lo & ~31; }

    for (int kb = kb_lo; kb <= qb + 15; kb += 32) {
        // ---- stage K,V 32x64 f16 tiles into this wave's LDS region ----
#if HAVE_ASYNC_LDS
        #pragma unroll
        for (int c = 0; c < 8; ++c) {
            const int flat = c * 32 + lane, key = flat >> 3, dq = flat & 7;
            const size_t gb = ((size_t)(kb + key) * hk + hkv) * HDIM + dq * 8;
            g2l_async_b128(kp + gb, &sK[wave][key][dq * 8]);
            g2l_async_b128(vp + gb, &sV[wave][key][dq * 8]);
        }
        WAIT_ASYNC0();
#else
        {
            uint4 tk[8], tv[8];
            #pragma unroll
            for (int c = 0; c < 8; ++c) {
                const int flat = c * 32 + lane, key = flat >> 3, dq = flat & 7;
                const size_t gb = ((size_t)(kb + key) * hk + hkv) * HDIM + dq * 8;
                tk[c] = *(const uint4*)(kp + gb);
                tv[c] = *(const uint4*)(vp + gb);
            }
            #pragma unroll
            for (int c = 0; c < 8; ++c) {
                const int flat = c * 32 + lane, key = flat >> 3, dq = flat & 7;
                *(uint4*)&sK[wave][key][dq * 8] = tk[c];
                *(uint4*)&sV[wave][key][dq * 8] = tv[c];
            }
        }
        asm volatile("s_wait_dscnt 0x0" ::: "memory");
#endif

        // ---- scores: 16x32 = two 16x16 accumulators, K-dim 64 = 2x32 ----
        v8f sc0, sc1;
        #pragma unroll
        for (int r = 0; r < 8; ++r) { sc0[r] = 0.0f; sc1[r] = 0.0f; }
        {
            v16h b0, b1;
            #pragma unroll
            for (int j = 0; j < 8; ++j) {          // key half 0
                const int dk = lg * 16 + 2 * j;
                b0[2 * j]     = sK[wave][l15][dk];
                b0[2 * j + 1] = sK[wave][l15][dk + 1];
                b1[2 * j]     = sK[wave][l15][32 + dk];
                b1[2 * j + 1] = sK[wave][l15][32 + dk + 1];
            }
            sc0 = __builtin_amdgcn_wmma_f32_16x16x32_f16(false, qa0, false, b0, (short)0, sc0, false, false);
            sc0 = __builtin_amdgcn_wmma_f32_16x16x32_f16(false, qa1, false, b1, (short)0, sc0, false, false);
            #pragma unroll
            for (int j = 0; j < 8; ++j) {          // key half 1
                const int dk = lg * 16 + 2 * j;
                b0[2 * j]     = sK[wave][16 + l15][dk];
                b0[2 * j + 1] = sK[wave][16 + l15][dk + 1];
                b1[2 * j]     = sK[wave][16 + l15][32 + dk];
                b1[2 * j + 1] = sK[wave][16 + l15][32 + dk + 1];
            }
            sc1 = __builtin_amdgcn_wmma_f32_16x16x32_f16(false, qa0, false, b0, (short)0, sc1, false, false);
            sc1 = __builtin_amdgcn_wmma_f32_16x16x32_f16(false, qa1, false, b1, (short)0, sc1, false, false);
        }

        // ---- masked online softmax (C-layout rows, 16-lane butterflies) ----
        #pragma unroll
        for (int r = 0; r < 8; ++r) {
            const int im = qb + ((lane < 16) ? r : r + 8);
            const int j0 = kb + l15, j1 = j0 + 16;
            const bool ok0 = (j0 <= im) && (window <= 0 || (im - j0) < window);
            const bool ok1 = (j1 <= im) && (window <= 0 || (im - j1) < window);
            float s0 = ok0 ? sc0[r] : -1e30f;
            float s1 = ok1 ? sc1[r] : -1e30f;
            float mx = fmaxf(s0, s1);
            #pragma unroll
            for (int d = 8; d; d >>= 1) mx = fmaxf(mx, __shfl_xor(mx, d, 32));
            const float mnew = fmaxf(mrow[r], mx);
            float p0 = ok0 ? __expf(s0 - mnew) : 0.0f;
            float p1 = ok1 ? __expf(s1 - mnew) : 0.0f;
            float ps = p0 + p1;
            #pragma unroll
            for (int d = 8; d; d >>= 1) ps += __shfl_xor(ps, d, 32);
            const float scl = __expf(mrow[r] - mnew);
            lrow[r] = lrow[r] * scl + ps;
            mrow[r] = mnew;
            #pragma unroll
            for (int t = 0; t < 4; ++t) o[t][r] *= scl;
            const int rowm = (lane < 16) ? r : r + 8;
            sP[wave][rowm][l15]      = (_Float16)p0;
            sP[wave][rowm][l15 + 16] = (_Float16)p1;
        }
        asm volatile("s_wait_dscnt 0x0" ::: "memory");

        // ---- O += P V : P as A fragment, V row-major B fragments ----
        v16h pa;
        #pragma unroll
        for (int j = 0; j < 8; ++j) {
            const int kk = lg * 16 + 2 * j;
            pa[2 * j]     = sP[wave][l15][kk];
            pa[2 * j + 1] = sP[wave][l15][kk + 1];
        }
        #pragma unroll
        for (int t = 0; t < 4; ++t) {
            v16h vb;
            #pragma unroll
            for (int j = 0; j < 8; ++j) {
                const int kk = lg * 16 + 2 * j;   // key index
                vb[2 * j]     = sV[wave][kk][t * 16 + l15];
                vb[2 * j + 1] = sV[wave][kk + 1][t * 16 + l15];
            }
            o[t] = __builtin_amdgcn_wmma_f32_16x16x32_f16(false, pa, false, vb, (short)0, o[t], false, false);
        }
    }

    // epilogue: O / l -> f16 concat attention buffer
    #pragma unroll
    for (int t = 0; t < 4; ++t)
        #pragma unroll
        for (int r = 0; r < 8; ++r) {
            const int rowm = (lane < 16) ? r : r + 8;
            out[(size_t)(qb + rowm) * out_ld + col_off + h * HDIM + t * 16 + l15] =
                (_Float16)(o[t][r] / lrow[r]);
        }
}

// ---------------------------------------------------------------------------
// Host-side orchestration
// ---------------------------------------------------------------------------
extern "C" void kernel_launch(void* const* d_in, const int* in_sizes, int n_in,
                              void* d_out, int out_size, void* d_ws, size_t ws_size,
                              hipStream_t stream) {
    (void)in_sizes; (void)n_in; (void)out_size; (void)ws_size;

    const float* x   = (const float*)d_in[0];
    const int*   pos = (const int*)  d_in[1];
    const float* w_in[7] = { (const float*)d_in[2], (const float*)d_in[3],
                             (const float*)d_in[4], (const float*)d_in[5],
                             (const float*)d_in[6], (const float*)d_in[7],
                             (const float*)d_in[10] };
    const float* q_norm_w = (const float*)d_in[8];
    const float* k_norm_w = (const float*)d_in[9];
    float* out = (float*)d_out;

    char* ws = (char*)d_ws;
    _Float16* xh      = (_Float16*)(ws + 0);         // 2048*2048*2
    _Float16* woh     = (_Float16*)(ws + 8388608);   // 2048*2048*2
    _Float16* wqh_swa = (_Float16*)(ws + 16777216);  // 2048*1536*2
    _Float16* wkh_swa = (_Float16*)(ws + 23068672);  // 2048* 384*2
    _Float16* wvh_swa = (_Float16*)(ws + 24641536);
    _Float16* wqh_hsa = (_Float16*)(ws + 26214400);  // 2048* 512*2
    _Float16* wkh_hsa = (_Float16*)(ws + 28311552);  // 2048* 128*2
    _Float16* wvh_hsa = (_Float16*)(ws + 28835840);
    _Float16* q_swa   = (_Float16*)(ws + 29360128);  // 2048*1536*2
    _Float16* k_swa   = (_Float16*)(ws + 35651584);  // 2048* 384*2
    _Float16* v_swa   = (_Float16*)(ws + 37224448);
    _Float16* q_hsa   = (_Float16*)(ws + 38797312);  // 2048* 512*2
    _Float16* k_hsa   = (_Float16*)(ws + 40894464);  // 2048* 128*2
    _Float16* v_hsa   = (_Float16*)(ws + 41418752);
    _Float16* attn    = (_Float16*)(ws + 41943040);  // 2048*2048*2  (end 50331648)

    const dim3 blk256(256), blk128(128);

    // --- bulk fp32 -> f16 conversions (x, all weights) ---
    _Float16* w_h[7] = { wqh_swa, wkh_swa, wvh_swa, wqh_hsa, wkh_hsa, wvh_hsa, woh };
    const int w_n[7] = { 2048*1536, 2048*384, 2048*384, 2048*512, 2048*128, 2048*128, 2048*2048 };
    cvt_f32_f16<<<dim3((2048*2048/4 + 255)/256), blk256, 0, stream>>>(x, xh, 2048*2048/4);
    for (int i = 0; i < 7; ++i)
        cvt_f32_f16<<<dim3((w_n[i]/4 + 255)/256), blk256, 0, stream>>>(w_in[i], w_h[i], w_n[i]/4);

    // --- QKV projections (M=2048, K=2048), f16 out ---
    gemm_h_wmma<<<dim3(1536/128, 16), blk256, 0, stream>>>(xh, 2048, wqh_swa, 1536, q_swa, 1536, 2048, 1);
    gemm_h_wmma<<<dim3( 384/128, 16), blk256, 0, stream>>>(xh, 2048, wkh_swa,  384, k_swa,  384, 2048, 1);
    gemm_h_wmma<<<dim3( 384/128, 16), blk256, 0, stream>>>(xh, 2048, wvh_swa,  384, v_swa,  384, 2048, 1);
    gemm_h_wmma<<<dim3( 512/128, 16), blk256, 0, stream>>>(xh, 2048, wqh_hsa,  512, q_hsa,  512, 2048, 1);
    gemm_h_wmma<<<dim3( 128/128, 16), blk256, 0, stream>>>(xh, 2048, wkh_hsa,  128, k_hsa,  128, 2048, 1);
    gemm_h_wmma<<<dim3( 128/128, 16), blk256, 0, stream>>>(xh, 2048, wvh_hsa,  128, v_hsa,  128, 2048, 1);

    // --- RMSNorm (+RoPE on SWA branch), in place on f16 ---
    norm_rope_h<<<dim3(S_LEN * 24 / 4), blk128, 0, stream>>>(q_swa, q_norm_w, pos, 24, 1);
    norm_rope_h<<<dim3(S_LEN *  6 / 4), blk128, 0, stream>>>(k_swa, k_norm_w, pos,  6, 1);
    norm_rope_h<<<dim3(S_LEN *  8 / 4), blk128, 0, stream>>>(q_hsa, q_norm_w, pos,  8, 0);
    norm_rope_h<<<dim3(S_LEN *  2 / 4), blk128, 0, stream>>>(k_hsa, k_norm_w, pos,  2, 0);

    // --- attention: SWA (window 512) + HSA (causal) into f16 concat buffer ---
    attn_h_wmma<<<dim3(24 * 128 / 4), blk128, 0, stream>>>(q_swa, k_swa, v_swa, attn, 24, 6, 512,    0, 2048);
    attn_h_wmma<<<dim3( 8 * 128 / 4), blk128, 0, stream>>>(q_hsa, k_hsa, v_hsa, attn,  8, 2,   0, 1536, 2048);

    // --- output projection: attn (f16) @ wo (f16) -> f32 d_out ---
    gemm_h_wmma<<<dim3(16, 16), blk256, 0, stream>>>(attn, 2048, woh, 2048, out, 2048, 2048, 0);
}
